// GRUIEncoder_36378372997185
// MI455X (gfx1250) — matrix operand order repaired and checked by reading, not compiled, not run on hardware
//
#include <hip/hip_runtime.h>
#include <hip/hip_bf16.h>

#define BB 256
#define TT 512
#define DD 128
#define HH 256

typedef __bf16 bf16_t;
typedef __attribute__((ext_vector_type(16))) __bf16 v16bf;
typedef __attribute__((ext_vector_type(8)))  __bf16 v8bf;
typedef __attribute__((ext_vector_type(8)))  float  v8f;
typedef __attribute__((ext_vector_type(4)))  float  v4f;
typedef __attribute__((ext_vector_type(4)))  int    v4i_t;
typedef __attribute__((address_space(3))) v4i_t*    lds_v4i_p;

// LDS layout (bytes from dynamic-shared base)
#define L_HS    0        // [16][256] f32 h state      16KB
#define L_HB    16384    // [16][256] bf16 h            8KB
#define L_RHB   24576    // [16][256] bf16 r*h          8KB
#define L_MUS   32768    // [16][256] bf16 mu           8KB
#define L_HHS   40960    // [16][256] bf16 h_hat        8KB
#define L_GS    49152    // [4][16][256] bf16 gates    32KB
#define L_WHH   81920    // packed Whh bf16           128KB
#define L_TOTAL 212992

__device__ inline char* lds_base() {
    extern __shared__ char smem[];
    return smem;
}

// Pin the schedule at tile boundaries (prevents cross-tile load hoisting).
__device__ inline void sched_fence() {
#if __has_builtin(__builtin_amdgcn_sched_barrier)
    __builtin_amdgcn_sched_barrier(0);
#else
    asm volatile("" ::: "memory");
#endif
}

// ---------------------------------------------------------------------------
// Packed B-fragment layout for V_WMMA_F32_16X16X32_BF16.
// B tile is 32(K) x 16(N) bf16 = 1KB block. Within a block:
//   lane l<16  : col n=l,    K = 0..7, 16..23  (16 bf16, 32B contiguous)
//   lane l>=16 : col n=l-16, K = 8..15, 24..31
// ---------------------------------------------------------------------------
__device__ __host__ inline size_t packed_off(int k, int c, int kchunks) {
    int nt = c >> 4, nl = c & 15;
    int kc = k >> 5, kl = k & 31;
    int lane = ((kl >> 3) & 1) * 16 + nl;
    int slot = (kl & 7) + ((kl >> 4) << 3);
    return ((size_t)(nt * kchunks + kc)) * 512 + (size_t)lane * 16 + slot;
}

// ws layout (bytes)
#define WS_WP    0          // [128][1024] bf16 blocked, KCH=4  (256KB)
#define WS_WHRM  262144     // [256][512]  bf16 blocked, KCH=8  (256KB)
#define WS_WHH   524288     // [256][256]  bf16 blocked, KCH=8  (128KB)
#define WS_BP    655360     // [1024] f32 biases [b_td|b_r|b_mu|b_h]

// ---------------------------------------------------------------------------
// Kernel 0: convert fp32 weights to bf16 and scatter into blocked layout.
// ---------------------------------------------------------------------------
__global__ void pack_weights_kernel(const float* __restrict__ Wmu,
                                    const float* __restrict__ bmu,
                                    const float* __restrict__ Wr,
                                    const float* __restrict__ br,
                                    const float* __restrict__ Wh,
                                    const float* __restrict__ bh,
                                    const float* __restrict__ Wtd,
                                    const float* __restrict__ btd,
                                    bf16_t* __restrict__ Wp,
                                    bf16_t* __restrict__ Whrm,
                                    bf16_t* __restrict__ Whh,
                                    float*  __restrict__ bp) {
    const int NWP   = DD * 1024;
    const int NWHRM = HH * 512;
    const int NWHH  = HH * HH;
    const int TOTAL = NWP + NWHRM + NWHH + 1024;
    for (int i = blockIdx.x * blockDim.x + threadIdx.x; i < TOTAL;
         i += gridDim.x * blockDim.x) {
        if (i < NWP) {
            int k = i / 1024, c = i % 1024;
            float v;
            if (c < 256)      v = Wtd[k * HH + c];
            else if (c < 512) v = Wr [(HH + k) * HH + (c - 256)];
            else if (c < 768) v = Wmu[(HH + k) * HH + (c - 512)];
            else              v = Wh [(HH + k) * HH + (c - 768)];
            Wp[packed_off(k, c, 4)] = (bf16_t)v;
        } else if (i < NWP + NWHRM) {
            int j = i - NWP;
            int k = j / 512, c = j % 512;
            float v = (c < 256) ? Wr[k * HH + c] : Wmu[k * HH + (c - 256)];
            Whrm[packed_off(k, c, 8)] = (bf16_t)v;
        } else if (i < NWP + NWHRM + NWHH) {
            int j = i - NWP - NWHRM;
            int k = j / 256, c = j % 256;
            Whh[packed_off(k, c, 8)] = (bf16_t)Wh[k * HH + c];
        } else {
            int c = i - NWP - NWHRM - NWHH;
            float v;
            if (c < 256)      v = btd[c];
            else if (c < 512) v = br [c - 256];
            else if (c < 768) v = bmu[c - 512];
            else              v = bh [c - 768];
            bp[c] = v;
        }
    }
}

// ---------------------------------------------------------------------------
// Fragment helpers
// ---------------------------------------------------------------------------
__device__ inline v16bf make_afrag_f32(const float* __restrict__ p) {
    v4f a0 = *(const v4f*)(p + 0);
    v4f a1 = *(const v4f*)(p + 4);
    v4f b0 = *(const v4f*)(p + 16);
    v4f b1 = *(const v4f*)(p + 20);
    v16bf r;
#pragma unroll
    for (int i = 0; i < 4; ++i) {
        r[i]      = (bf16_t)a0[i];
        r[4 + i]  = (bf16_t)a1[i];
        r[8 + i]  = (bf16_t)b0[i];
        r[12 + i] = (bf16_t)b1[i];
    }
    return r;
}

__device__ inline v16bf make_afrag_lds(const bf16_t* base) {
    v8bf lo = *(const v8bf*)(base);
    v8bf hi = *(const v8bf*)(base + 16);
    v16bf r;
#pragma unroll
    for (int i = 0; i < 8; ++i) { r[i] = lo[i]; r[8 + i] = hi[i]; }
    return r;
}

__device__ inline v16bf load_bfrag(const bf16_t* block, int lane) {
    return *(const v16bf*)(block + lane * 16);
}

__device__ inline v8f wmma_bf16(v16bf a, v16bf b, v8f c) {
    return __builtin_amdgcn_wmma_f32_16x16x32_bf16(false, a, false, b, (short)0, c,
                                                   false, false);
}

__device__ inline float sigm(float z) { return 1.0f / (1.0f + __expf(-z)); }

// ---------------------------------------------------------------------------
// Phase functions: __noinline__ creates hard register-pressure boundaries so
// loop-invariant addresses are NOT kept live across the whole t-loop.
// ---------------------------------------------------------------------------

// Phase A: gates = act((x|delta) @ Wp + b) -> gS   (16x128 @ 128x1024)
__device__ __attribute__((noinline))
void do_phaseA(const float* __restrict__ srcA, const bf16_t* __restrict__ Wp,
               const float* __restrict__ bp, int t, int b0) {
    char* smem = lds_base();
    bf16_t* gS = (bf16_t*)(smem + L_GS);
    const int tid = threadIdx.x;
    const int w = tid >> 5, lane = tid & 31, col = lane & 15, half = lane >> 4;
    const int m = col, startK = half * 8;

    v16bf afr[4];
#pragma unroll
    for (int kc = 0; kc < 4; ++kc) {
        const float* p = srcA + ((size_t)(b0 + m) * TT + t) * DD + kc * 32 + startK;
        afr[kc] = make_afrag_f32(p);
        __builtin_prefetch(p + DD, 0, 1);   // next timestep (speculative)
        sched_fence();
    }
#pragma unroll
    for (int jt = 0; jt < 4; ++jt) {
        const int gtile = w * 4 + jt;
        v8f acc = {};
#pragma unroll
        for (int kc = 0; kc < 4; ++kc) {
            v16bf bfr = load_bfrag(Wp + ((size_t)(gtile * 4 + kc)) * 512, lane);
            acc = wmma_bf16(afr[kc], bfr, acc);
        }
        const int comp  = gtile >> 4;                 // 0:beta 1:gx_r 2:gx_mu 3:gx_h
        const int ccol  = (gtile & 15) * 16 + col;
        const float bias = bp[gtile * 16 + col];
#pragma unroll
        for (int vr = 0; vr < 8; ++vr) {
            const int mrow = half * 8 + vr;
            float z = acc[vr] + bias;
            if (comp == 0) z = __expf(-fmaxf(z, 0.0f));   // beta
            gS[comp * 4096 + mrow * 256 + ccol] = (bf16_t)z;
        }
        sched_fence();
    }
}

// Phase C: [r|mu] = sigmoid(h @ Whrm + gx); r-half also writes r*h.
__device__ __attribute__((noinline))
void do_phaseC(const bf16_t* __restrict__ Whrm) {
    char* smem = lds_base();
    float*  hS  = (float*)(smem + L_HS);
    bf16_t* hB  = (bf16_t*)(smem + L_HB);
    bf16_t* rhB = (bf16_t*)(smem + L_RHB);
    bf16_t* muS = (bf16_t*)(smem + L_MUS);
    bf16_t* gS  = (bf16_t*)(smem + L_GS);
    const int tid = threadIdx.x;
    const int w = tid >> 5, lane = tid & 31, col = lane & 15, half = lane >> 4;
    const int m = col, startK = half * 8;

#pragma unroll
    for (int jt = 0; jt < 2; ++jt) {
        const int tile = w * 2 + jt;
        v8f acc = {};
#pragma unroll
        for (int kc = 0; kc < 8; ++kc) {
            v16bf a = make_afrag_lds(hB + m * 256 + kc * 32 + startK);
            v16bf f = load_bfrag(Whrm + ((size_t)(tile * 8 + kc)) * 512, lane);
            acc = wmma_bf16(a, f, acc);
        }
        const int n = tile * 16 + col;        // [0,512)
        if (n < 256) {                        // r half -> r*h
#pragma unroll
            for (int vr = 0; vr < 8; ++vr) {
                const int mrow = half * 8 + vr;
                float z = acc[vr] + (float)gS[1 * 4096 + mrow * 256 + n];
                float r = sigm(z);
                rhB[mrow * 256 + n] = (bf16_t)(r * hS[mrow * 256 + n]);
            }
        } else {                              // mu half
            const int nm = n - 256;
#pragma unroll
            for (int vr = 0; vr < 8; ++vr) {
                const int mrow = half * 8 + vr;
                float z = acc[vr] + (float)gS[2 * 4096 + mrow * 256 + nm];
                muS[mrow * 256 + nm] = (bf16_t)sigm(z);
            }
        }
        sched_fence();
    }
}

// Phase D: h_hat = tanh((r*h) @ Whh + gx_h)  (Whh resident in LDS)
__device__ __attribute__((noinline))
void do_phaseD() {
    char* smem = lds_base();
    bf16_t* rhB  = (bf16_t*)(smem + L_RHB);
    bf16_t* hhS  = (bf16_t*)(smem + L_HHS);
    bf16_t* gS   = (bf16_t*)(smem + L_GS);
    bf16_t* WhhL = (bf16_t*)(smem + L_WHH);
    const int tid = threadIdx.x;
    const int w = tid >> 5, lane = tid & 31, col = lane & 15, half = lane >> 4;
    const int m = col, startK = half * 8;

    v8f acc = {};
#pragma unroll
    for (int kc = 0; kc < 8; ++kc) {
        v16bf a   = make_afrag_lds(rhB + m * 256 + kc * 32 + startK);
        v16bf bfr = load_bfrag(WhhL + ((size_t)(w * 8 + kc)) * 512, lane);
        acc = wmma_bf16(a, bfr, acc);
    }
    const int n = w * 16 + col;
#pragma unroll
    for (int vr = 0; vr < 8; ++vr) {
        const int mrow = half * 8 + vr;
        float z = acc[vr] + (float)gS[3 * 4096 + mrow * 256 + n];
        hhS[mrow * 256 + n] = (bf16_t)tanhf(z);
    }
}

// ---------------------------------------------------------------------------
// Kernel 1: fused gate-precompute + sequential scan.
// grid = 16 blocks (16 batch rows each), block = 512 threads = 16 waves.
// ---------------------------------------------------------------------------
__global__ void __launch_bounds__(512)
__attribute__((amdgpu_waves_per_eu(4, 4)))
gru_scan_kernel(const float* __restrict__ x,
                const float* __restrict__ delta,
                const bf16_t* __restrict__ Wp,
                const bf16_t* __restrict__ Whrm,
                const bf16_t* __restrict__ Whh,
                const float* __restrict__ bp,
                float* __restrict__ out) {
    char* smem = lds_base();
    float*  hS   = (float*)(smem + L_HS);
    bf16_t* hB   = (bf16_t*)(smem + L_HB);
    bf16_t* muS  = (bf16_t*)(smem + L_MUS);
    bf16_t* hhS  = (bf16_t*)(smem + L_HHS);
    bf16_t* gS   = (bf16_t*)(smem + L_GS);
    bf16_t* WhhL = (bf16_t*)(smem + L_WHH);

    const int tid = threadIdx.x;
    const int w   = tid >> 5;
    const int b0  = blockIdx.x * 16;

    // ---- preload packed Whh into LDS (async-to-LDS path) ----
    {
        const v4i_t* src = (const v4i_t*)Whh;   // 16384 x 16B
        v4i_t*       dst = (v4i_t*)WhhL;
#if __has_builtin(__builtin_amdgcn_global_load_async_to_lds_b128)
#pragma unroll 4
        for (int i = tid; i < 16384; i += 512) {
            __builtin_amdgcn_global_load_async_to_lds_b128(
                (v4i_t*)(src + i), (lds_v4i_p)(dst + i), 0, 0);
        }
#if __has_builtin(__builtin_amdgcn_s_wait_asynccnt)
        __builtin_amdgcn_s_wait_asynccnt(0);
#else
        asm volatile("s_wait_asynccnt 0" ::: "memory");
#endif
#else
#pragma unroll 4
        for (int i = tid; i < 16384; i += 512) dst[i] = src[i];
#endif
    }

    // zero h
#pragma unroll
    for (int i = 0; i < 8; ++i) hS[i * 512 + tid] = 0.0f;
    __syncthreads();

    // waves 0..3 compute beta (delta A-matrix), waves 4..15 gx (x A-matrix)
    const float* srcA = (w < 4) ? delta : x;

    for (int t = 0; t < TT; ++t) {
        do_phaseA(srcA, Wp, bp, t, b0);
        __syncthreads();

        // Phase B: h = beta * h
#pragma unroll
        for (int i = 0; i < 8; ++i) {
            const int idx = i * 512 + tid;
            float h = hS[idx] * (float)gS[idx];   // gS comp 0
            hS[idx] = h;
            hB[idx] = (bf16_t)h;
        }
        __syncthreads();

        do_phaseC(Whrm);
        __syncthreads();

        do_phaseD();
        __syncthreads();

        // Phase E: h = (1-mu)*h + mu*h_hat
#pragma unroll
        for (int i = 0; i < 8; ++i) {
            const int idx = i * 512 + tid;
            float mu = (float)muS[idx];
            float hh = (float)hhS[idx];
            float h  = hS[idx];
            hS[idx] = (1.0f - mu) * h + mu * hh;
        }
        __syncthreads();
    }

    // write h_T (fp32) for this block's 16 rows
#pragma unroll
    for (int i = 0; i < 8; ++i) {
        const int idx = i * 512 + tid;
        const int row = idx >> 8, c = idx & 255;
        out[(size_t)(b0 + row) * HH + c] = hS[idx];
    }
}

// ---------------------------------------------------------------------------
extern "C" void kernel_launch(void* const* d_in, const int* in_sizes, int n_in,
                              void* d_out, int out_size, void* d_ws, size_t ws_size,
                              hipStream_t stream) {
    const float* x     = (const float*)d_in[0];
    const float* delta = (const float*)d_in[1];
    const float* Wmu   = (const float*)d_in[2];
    const float* bmu   = (const float*)d_in[3];
    const float* Wr    = (const float*)d_in[4];
    const float* br    = (const float*)d_in[5];
    const float* Wh    = (const float*)d_in[6];
    const float* bh    = (const float*)d_in[7];
    const float* Wtd   = (const float*)d_in[8];
    const float* btd   = (const float*)d_in[9];

    char* ws = (char*)d_ws;
    bf16_t* Wp   = (bf16_t*)(ws + WS_WP);
    bf16_t* Whrm = (bf16_t*)(ws + WS_WHRM);
    bf16_t* Whh  = (bf16_t*)(ws + WS_WHH);
    float*  bp   = (float*)(ws + WS_BP);

    pack_weights_kernel<<<256, 256, 0, stream>>>(Wmu, bmu, Wr, br, Wh, bh, Wtd, btd,
                                                 Wp, Whrm, Whh, bp);

    gru_scan_kernel<<<16, 512, L_TOTAL, stream>>>(x, delta, Wp, Whrm, Whh, bp,
                                                  (float*)d_out);
}
